// QuantumKernel_65481071403386
// MI455X (gfx1250) — matrix operand order repaired and checked by reading, not compiled, not run on hardware
//
#include <hip/hip_runtime.h>

// ---------------------------------------------------------------------------
// RBF kernel exp(-gamma*||x-y||^2) via bf16x3 split GEMM on CDNA5 WMMA.
//   s = Xhi*Yhi^T + Xhi*Ylo^T + Xlo*Yhi^T   (f32 accumulate, ~f32 accuracy)
//   out = exp(-gamma * max(0, x2 + y2 - 2 s))
// Double-buffered LDS staging with gfx1250 async global->LDS copies
// (ASYNCcnt), one workgroup barrier per K tile, non-temporal output stores.
// ---------------------------------------------------------------------------

typedef __bf16 bf16_t;
typedef __attribute__((ext_vector_type(8)))  __bf16 v8bf;
typedef __attribute__((ext_vector_type(16))) __bf16 v16bf;
typedef __attribute__((ext_vector_type(8)))  float  v8f;

#define NROWS 8192   // rows of X (and of Y)
#define DDIM  512    // feature dim

#define BM   128     // output tile M per block
#define BN   128     // output tile N per block
#define KC   32      // K tile staged in LDS
#define LDSS 40      // padded LDS row stride in bf16 (80B: 16B aligned, conflict-free)

#define ARR_ELEMS (BM * LDSS)          // one staged matrix (bf16 elems)
#define ARR_BYTES (ARR_ELEMS * 2)
#define BUF_ELEMS (4 * ARR_ELEMS)      // Xhi,Xlo,Yhi,Ylo
#define BUF_BYTES (4 * ARR_BYTES)

// ---- split each f32 into bf16 hi + bf16 lo, and compute row norms ----------
__global__ void split_norm_kernel(const float* __restrict__ src,
                                  bf16_t* __restrict__ hi,
                                  bf16_t* __restrict__ lo,
                                  float* __restrict__ norm) {
    __shared__ float red[256];
    const int row = blockIdx.x;
    const int t = threadIdx.x;
    const float* s = src + (size_t)row * DDIM;
    bf16_t* h = hi + (size_t)row * DDIM;
    bf16_t* l = lo + (size_t)row * DDIM;
    float acc = 0.0f;
    for (int i = t; i < DDIM; i += 256) {
        float x = s[i];
        acc += x * x;
        bf16_t xh = (bf16_t)x;                 // RNE to bf16
        bf16_t xl = (bf16_t)(x - (float)xh);   // residual in bf16
        h[i] = xh;
        l[i] = xl;
    }
    red[t] = acc;
    __syncthreads();
    for (int off = 128; off > 0; off >>= 1) {
        if (t < off) red[t] += red[t + off];
        __syncthreads();
    }
    if (t == 0) norm[row] = red[0];
}

__device__ __forceinline__ v16bf cat8(v8bf a, v8bf b) {
    return __builtin_shufflevector(a, b, 0, 1, 2, 3, 4, 5, 6, 7,
                                         8, 9, 10, 11, 12, 13, 14, 15);
}

// gfx1250 async global->LDS 16B copy (tracked by ASYNCcnt, no VGPR roundtrip)
__device__ __forceinline__ void async_b128(unsigned lds_addr, const bf16_t* g) {
    asm volatile("global_load_async_to_lds_b128 %0, %1, off"
                 :: "v"(lds_addr), "v"((unsigned long long)(size_t)g)
                 : "memory");
}

__device__ __forceinline__ void wait_async0() {
#if __has_builtin(__builtin_amdgcn_s_wait_asynccnt)
    __builtin_amdgcn_s_wait_asynccnt(0);
#else
    asm volatile("s_wait_asynccnt 0x0" ::: "memory");
#endif
}

// ---- main fused GEMM + RBF epilogue ----------------------------------------
__global__ __launch_bounds__(256)
void rbf_wmma_kernel(const bf16_t* __restrict__ Xhi, const bf16_t* __restrict__ Xlo,
                     const bf16_t* __restrict__ Yhi, const bf16_t* __restrict__ Ylo,
                     const float* __restrict__ x2g, const float* __restrict__ y2g,
                     const float* __restrict__ gamma_p,
                     float* __restrict__ out) {
    __shared__ bf16_t smem[2 * BUF_ELEMS];   // double-buffered Xhi|Xlo|Yhi|Ylo
    __shared__ float sx2[BM];
    __shared__ float sy2[BN];

    const int t  = threadIdx.x;
    const int bm = blockIdx.y * BM;
    const int bn = blockIdx.x * BN;

    if (t < BM) sx2[t] = x2g[bm + t];
    else        sy2[t - BM] = y2g[bn + (t - BM)];

    // 8 waves in a 4 (M) x 2 (N) grid; each wave owns 32(M) x 64(N)
    const int wave  = t >> 5;
    const int lane  = t & 31;
    const int h     = lane >> 4;   // lane half (0: lanes 0-15, 1: lanes 16-31)
    const int r     = lane & 15;   // row-within-16
    const int waveM = (wave >> 1) * 32;
    const int waveN = (wave & 1) * 64;

    // LDS byte address of smem (low 32 bits of the flat shared pointer)
    const unsigned smem_lds = (unsigned)(size_t)(&smem[0]);

    v8f acc[2][4];
    const v8f vz = {0.f, 0.f, 0.f, 0.f, 0.f, 0.f, 0.f, 0.f};
    #pragma unroll
    for (int mi = 0; mi < 2; ++mi)
        #pragma unroll
        for (int ni = 0; ni < 4; ++ni)
            acc[mi][ni] = vz;

    // stage one K-tile (4 arrays) into LDS buffer `buf` via async copies:
    // 128 rows x 4 16B-chunks = 512 chunks per array; 256 threads -> 2 each.
    auto stage = [&](int kt, int buf) {
        #pragma unroll
        for (int it = 0; it < 2; ++it) {
            int c   = t + it * 256;
            int row = c >> 2;
            int col = (c & 3) * 8;
            size_t gx = (size_t)(bm + row) * DDIM + kt + col;
            size_t gy = (size_t)(bn + row) * DDIM + kt + col;
            unsigned lb = smem_lds + (unsigned)(buf * BUF_BYTES + (row * LDSS + col) * 2);
            async_b128(lb + 0 * ARR_BYTES, Xhi + gx);
            async_b128(lb + 1 * ARR_BYTES, Xlo + gx);
            async_b128(lb + 2 * ARR_BYTES, Yhi + gy);
            async_b128(lb + 3 * ARR_BYTES, Ylo + gy);
        }
    };

    auto compute = [&](int buf) {
        const bf16_t* base = smem + (size_t)buf * BUF_ELEMS;
        const bf16_t* bXhi = base;
        const bf16_t* bXlo = base + 1 * ARR_ELEMS;
        const bf16_t* bYhi = base + 2 * ARR_ELEMS;
        const bf16_t* bYlo = base + 3 * ARR_ELEMS;

        // A fragments (16x32 bf16, ISA 7.12.2): lane(h,r) regs[0..7]=K 8h..8h+7,
        // regs[8..15]=K 16+8h..16+8h+7, row M=r.
        v16bf ahi[2], alo[2];
        #pragma unroll
        for (int mi = 0; mi < 2; ++mi) {
            const bf16_t* rp = bXhi + (waveM + mi * 16 + r) * LDSS;
            ahi[mi] = cat8(*(const v8bf*)(rp + 8 * h),
                           *(const v8bf*)(rp + 16 + 8 * h));
            const bf16_t* rl = bXlo + (waveM + mi * 16 + r) * LDSS;
            alo[mi] = cat8(*(const v8bf*)(rl + 8 * h),
                           *(const v8bf*)(rl + 16 + 8 * h));
        }
        // B fragments (32x16 bf16): lane(h,r) regs[0..7]=K 8h..8h+7, regs[8..15]=K 16+8h..16+8h+7 of column N=r.
        #pragma unroll
        for (int ni = 0; ni < 4; ++ni) {
            const bf16_t* rp = bYhi + (waveN + ni * 16 + r) * LDSS + 8 * h;
            v16bf bhi = cat8(*(const v8bf*)(rp), *(const v8bf*)(rp + 16));
            const bf16_t* rl = bYlo + (waveN + ni * 16 + r) * LDSS + 8 * h;
            v16bf blo = cat8(*(const v8bf*)(rl), *(const v8bf*)(rl + 16));
            #pragma unroll
            for (int mi = 0; mi < 2; ++mi) {
                acc[mi][ni] = __builtin_amdgcn_wmma_f32_16x16x32_bf16(
                    false, ahi[mi], false, bhi, (short)0, acc[mi][ni], false, false);
                acc[mi][ni] = __builtin_amdgcn_wmma_f32_16x16x32_bf16(
                    false, ahi[mi], false, blo, (short)0, acc[mi][ni], false, false);
                acc[mi][ni] = __builtin_amdgcn_wmma_f32_16x16x32_bf16(
                    false, alo[mi], false, bhi, (short)0, acc[mi][ni], false, false);
            }
        }
    };

    // software pipeline: prefetch tile k+1 (async) while computing tile k
    stage(0, 0);
    wait_async0();
    __syncthreads();

    int buf = 0;
    for (int kt = 0; kt < DDIM; kt += KC) {
        if (kt + KC < DDIM) stage(kt + KC, buf ^ 1);
        compute(buf);
        wait_async0();
        __syncthreads();
        buf ^= 1;
    }

    // Epilogue: C/D 16x16 f32 layout -> reg i, lane(h,r): M = 8h+i, N = r.
    // Non-temporal stores: output (256MB) is streamed once; keep the bf16
    // split operands (32MB) resident in the 192MB L2 instead.
    const float gamma = gamma_p[0];
    #pragma unroll
    for (int mi = 0; mi < 2; ++mi) {
        const int mloc = waveM + mi * 16 + 8 * h;
        #pragma unroll
        for (int ni = 0; ni < 4; ++ni) {
            const int nloc = waveN + ni * 16 + r;
            const float ysq = sy2[nloc];
            size_t ob = (size_t)(bm + mloc) * NROWS + bn + nloc;
            #pragma unroll
            for (int i = 0; i < 8; ++i) {
                float s  = acc[mi][ni][i];
                float sq = sx2[mloc + i] + ysq - 2.0f * s;
                sq = fmaxf(sq, 0.0f);
                __builtin_nontemporal_store(__expf(-gamma * sq),
                                            out + ob + (size_t)i * NROWS);
            }
        }
    }
}

extern "C" void kernel_launch(void* const* d_in, const int* in_sizes, int n_in,
                              void* d_out, int out_size, void* d_ws, size_t ws_size,
                              hipStream_t stream) {
    (void)in_sizes; (void)n_in; (void)out_size; (void)ws_size;
    const float* X     = (const float*)d_in[0];
    const float* Y     = (const float*)d_in[1];
    const float* gamma = (const float*)d_in[2];
    float* out = (float*)d_out;

    // workspace layout: 4 bf16 matrices (8 MB each) + 2 norm vectors
    char* ws = (char*)d_ws;
    const size_t mat_bytes = (size_t)NROWS * DDIM * sizeof(bf16_t);
    bf16_t* Xhi = (bf16_t*)(ws + 0 * mat_bytes);
    bf16_t* Xlo = (bf16_t*)(ws + 1 * mat_bytes);
    bf16_t* Yhi = (bf16_t*)(ws + 2 * mat_bytes);
    bf16_t* Ylo = (bf16_t*)(ws + 3 * mat_bytes);
    float*  x2  = (float*)(ws + 4 * mat_bytes);
    float*  y2  = x2 + NROWS;

    split_norm_kernel<<<NROWS, 256, 0, stream>>>(X, Xhi, Xlo, x2);
    split_norm_kernel<<<NROWS, 256, 0, stream>>>(Y, Yhi, Ylo, y2);

    dim3 grid(NROWS / BN, NROWS / BM);
    rbf_wmma_kernel<<<grid, 256, 0, stream>>>(Xhi, Xlo, Yhi, Ylo, x2, y2, gamma, out);
}
